// Decoder_attention_60662118088813
// MI455X (gfx1250) — matrix operand order repaired and checked by reading, not compiled, not run on hardware
//
#include <hip/hip_runtime.h>

#define B_   64
#define T_   256
#define H_   512
#define KDIM 512
#define V_   512
#define F_   4096
#define KK   1024     // combined inner dim ([x,h] concat)
#define NG   2048     // gate columns (4*H)
#define NWG  64
#define TPB  256
#define LDP  68       // LDS tile pitch (64 + pad)

typedef __attribute__((ext_vector_type(16))) __bf16 v16bf;
typedef __attribute__((ext_vector_type(8)))  __bf16 v8bf;
typedef __attribute__((ext_vector_type(8)))  float  v8f;

// ---------------- scalar helpers ----------------
__device__ inline __bf16 f2bf(float f) {
  union { float f; unsigned u; } v; v.f = f;
  unsigned r = v.u + 0x7FFFu + ((v.u >> 16) & 1u);   // round-to-nearest-even
  unsigned short h = (unsigned short)(r >> 16);
  __bf16 o; __builtin_memcpy(&o, &h, 2); return o;
}
__device__ inline float sigf(float x) { return 1.f / (1.f + __expf(-x)); }
__device__ inline float tanh_(float x) {
  x = fminf(fmaxf(x, -15.f), 15.f);
  float e = __expf(2.f * x);
  return (e - 1.f) / (e + 1.f);
}
__device__ inline float wred_sum(float v) {
  #pragma unroll
  for (int off = 16; off > 0; off >>= 1) v += __shfl_xor(v, off, 32);
  return v;
}
__device__ inline float wred_max(float v) {
  #pragma unroll
  for (int off = 16; off > 0; off >>= 1) v = fmaxf(v, __shfl_xor(v, off, 32));
  return v;
}

// ---------------- WMMA fragment loads (bf16 16x16x32, wave32) ----------------
// A (16x32, M x K), row-major X with ld = KK:
//   lanes 0-15 : row M=l,    K = kc+0..15
//   lanes16-31 : row M=l-16, K = kc+16..31
__device__ inline v16bf load_afrag(const __bf16* __restrict__ X, int m0, int kc, int lane) {
  const __bf16* p = X + (size_t)(m0 + (lane & 15)) * KK + kc + ((lane >> 4) << 4);
  v8bf lo = *(const v8bf*)p;
  v8bf hi = *(const v8bf*)(p + 8);
  v16bf a;
  #pragma unroll
  for (int i = 0; i < 8; ++i) { a[i] = lo[i]; a[i + 8] = hi[i]; }
  return a;
}
// B (32x16, K x N), weights packed as Wp[n][k] (k contiguous, ld = KK):
//   lanes 0-15 : col N=l,    K = kc+0..15  -> one contiguous 32B load
//   lanes16-31 : col N=l-16, K = kc+16..31
__device__ inline v16bf load_bfrag(const __bf16* __restrict__ Wp, int nb, int kc, int lane) {
  const __bf16* p = Wp + (size_t)(nb + (lane & 15)) * KK + kc + ((lane >> 4) << 4);
  return *(const v16bf*)p;
}
__device__ inline v8f wmma_bf16(v16bf a, v16bf b, v8f c) {
  return __builtin_amdgcn_wmma_f32_16x16x32_bf16(false, a, false, b, (short)0, c, false, false);
}

// 16(M) x 32(N) accumulation over K=1024 for one wave.
// Software-pipelined: fragments for chunk kc+32 are loaded before the WMMAs
// for chunk kc are issued, so s_wait_loadcnt retires a full iteration early.
__device__ inline void gemm_core(const __bf16* __restrict__ X, const __bf16* __restrict__ Wp,
                                 int nbase, int m0, int lane, v8f& a0, v8f& a1) {
  v16bf af  = load_afrag(X, m0, 0, lane);
  v16bf bf0 = load_bfrag(Wp, nbase, 0, lane);
  v16bf bf1 = load_bfrag(Wp, nbase + 16, 0, lane);
  #pragma unroll 4
  for (int kc = 0; kc < KK - 32; kc += 32) {
    v16bf an  = load_afrag(X, m0, kc + 32, lane);
    v16bf bn0 = load_bfrag(Wp, nbase, kc + 32, lane);
    v16bf bn1 = load_bfrag(Wp, nbase + 16, kc + 32, lane);
    __builtin_prefetch((const void*)(Wp + (size_t)(nbase + (lane & 15)) * KK + kc + 96), 0, 1);
    a0 = wmma_bf16(af, bf0, a0);
    a1 = wmma_bf16(af, bf1, a1);
    af = an; bf0 = bn0; bf1 = bn1;
  }
  a0 = wmma_bf16(af, bf0, a0);
  a1 = wmma_bf16(af, bf1, a1);
}

// ---------------- grid barrier (sense via generation counter) ----------------
__device__ inline void grid_barrier(unsigned* bar) {
  __syncthreads();
  if (threadIdx.x == 0) {
    __threadfence();
    unsigned* cnt = bar;
    unsigned* gen = bar + 1;
    unsigned g = __atomic_load_n(gen, __ATOMIC_ACQUIRE);
    if (atomicAdd(cnt, 1u) == (unsigned)(NWG - 1)) {
      __atomic_store_n(cnt, 0u, __ATOMIC_RELAXED);
      __atomic_fetch_add(gen, 1u, __ATOMIC_RELEASE);
    } else {
      while (__atomic_load_n(gen, __ATOMIC_ACQUIRE) == g)
        __builtin_amdgcn_s_sleep(2);
    }
    __threadfence();
  }
  __syncthreads();
}

// ---------------- fused LSTM tile: 64x64 GEMM + gates + cell update ----------------
// Columns are gate-interleaved: n = 4*u + g, so a 64-col tile owns 16 full units.
__device__ inline void lstm_tile(const __bf16* __restrict__ X, const __bf16* __restrict__ Wp,
                                 const float* __restrict__ bias, int nb,
                                 float* __restrict__ cbuf,
                                 __bf16* __restrict__ ha,   // write h at col 512+u
                                 __bf16* __restrict__ hb,   // write h at col u
                                 float* __restrict__ hf,    // optional fp32 h
                                 float* smem, int tid) {
  const int lane = tid & 31, w = tid >> 5;
  const int m0 = (w & 3) * 16;
  const int nw = (w >> 2) * 32;
  v8f a0 = {0.f,0.f,0.f,0.f,0.f,0.f,0.f,0.f}, a1 = a0;
  gemm_core(X, Wp, nb + nw, m0, lane, a0, a1);

  __syncthreads();                       // previous smem use complete
  const int nloc = lane & 15, mh = (lane >> 4) * 8;
  #pragma unroll
  for (int r = 0; r < 8; ++r) {
    smem[(m0 + mh + r) * LDP + nw + nloc]      = a0[r];
    smem[(m0 + mh + r) * LDP + nw + 16 + nloc] = a1[r];
  }
  __syncthreads();

  const int u0 = nb >> 2;
  for (int idx = tid; idx < B_ * 16; idx += TPB) {
    int b = idx & 63, ul = idx >> 6;
    float gi = smem[b * LDP + ul * 4 + 0] + bias[nb + ul * 4 + 0];
    float gf = smem[b * LDP + ul * 4 + 1] + bias[nb + ul * 4 + 1];
    float gg = smem[b * LDP + ul * 4 + 2] + bias[nb + ul * 4 + 2];
    float go = smem[b * LDP + ul * 4 + 3] + bias[nb + ul * 4 + 3];
    int u = u0 + ul;
    float co = cbuf[b * H_ + u];
    float cn = sigf(gf) * co + sigf(gi) * tanh_(gg);
    float hn = sigf(go) * tanh_(cn);
    cbuf[b * H_ + u] = cn;
    __bf16 hv = f2bf(hn);
    ha[(size_t)b * KK + H_ + u] = hv;
    hb[(size_t)b * KK + u] = hv;
    if (hf) hf[b * KDIM + u] = hn;
  }
  __syncthreads();
}

// ---------------- output-projection tile: 64x64 GEMM, direct store ----------------
__device__ inline void proj_tile(const __bf16* __restrict__ X, const __bf16* __restrict__ Wp,
                                 const float* __restrict__ bias, int nb,
                                 float* __restrict__ outp, int tid) {
  const int lane = tid & 31, w = tid >> 5;
  const int m0 = (w & 3) * 16;
  const int nw = (w >> 2) * 32;
  v8f a0 = {0.f,0.f,0.f,0.f,0.f,0.f,0.f,0.f}, a1 = a0;
  gemm_core(X, Wp, nb + nw, m0, lane, a0, a1);
  const int nloc = lane & 15, mh = (lane >> 4) * 8;
  #pragma unroll
  for (int r = 0; r < 8; ++r) {
    int b = m0 + mh + r;
    int c0 = nb + nw + nloc, c1c = c0 + 16;
    outp[(size_t)b * T_ * F_ + c0]  = a0[r] + bias[c0];
    outp[(size_t)b * T_ * F_ + c1c] = a1[r] + bias[c1c];
  }
}

// ---------------- packing / init kernels ----------------
__global__ void pack_lstm_kernel(const float* __restrict__ Wih, const float* __restrict__ Whh,
                                 const float* __restrict__ bih, const float* __restrict__ bhh,
                                 __bf16* __restrict__ Wp, float* __restrict__ bp) {
  size_t idx = (size_t)blockIdx.x * TPB + threadIdx.x;
  if (idx < (size_t)NG * KK) {
    int n = (int)(idx >> 10), k = (int)(idx & 1023);
    int u = n >> 2, g = n & 3;
    int row = g * H_ + u;
    float v = (k < H_) ? Wih[(size_t)row * H_ + k] : Whh[(size_t)row * H_ + (k - H_)];
    Wp[idx] = f2bf(v);
  }
  if (idx < NG) {
    int u = (int)(idx >> 2), g = (int)(idx & 3);
    bp[idx] = bih[g * H_ + u] + bhh[g * H_ + u];
  }
}

__global__ void pack_out_kernel(const float* __restrict__ Wo, const float* __restrict__ bo,
                                __bf16* __restrict__ Wp, float* __restrict__ bp) {
  size_t idx = (size_t)blockIdx.x * TPB + threadIdx.x;
  if (idx < (size_t)F_ * KK) Wp[idx] = f2bf(Wo[idx]);
  if (idx < F_) bp[idx] = bo[idx];
}

__global__ void init_state_kernel(const float* __restrict__ values,
                                  __bf16* xh1, __bf16* xh2, __bf16* xho,
                                  float* c1, float* c2, unsigned* bar) {
  int idx = blockIdx.x * TPB + threadIdx.x;     // 0 .. B_*KK-1
  if (idx < B_ * KK) {
    int b = idx >> 10, k = idx & 1023;
    __bf16 z = f2bf(0.f);
    xh1[idx] = (k < H_) ? f2bf(values[(size_t)b * T_ * V_ + k]) : z;  // ctx0 = values[:,0,:], h1=0
    xh1[B_ * KK + idx] = z;
    xh2[idx] = z;  xh2[B_ * KK + idx] = z;
    xho[idx] = z;  xho[B_ * KK + idx] = z;
  }
  if (idx < B_ * H_) { c1[idx] = 0.f; c2[idx] = 0.f; }
  if (idx < 16) bar[idx] = 0u;
}

// ---------------- persistent decoder kernel ----------------
__global__ __launch_bounds__(TPB) void decoder_persistent(
    const float* __restrict__ keys, const float* __restrict__ values,
    const __bf16* __restrict__ W1p, const __bf16* __restrict__ W2p, const __bf16* __restrict__ Wop,
    const float* __restrict__ b1p, const float* __restrict__ b2p, const float* __restrict__ bop,
    __bf16* xh1, __bf16* xh2, __bf16* xho,
    float* c1, float* c2, float* h2f, float* energy,
    unsigned* bar, float* out) {
  __shared__ float smem[B_ * LDP];
  __shared__ float sattn[T_];
  __shared__ float sred[8];
  __shared__ float sbc[2];
  const int tid = threadIdx.x;
  const int wg = blockIdx.x;
  const int lane = tid & 31;
  const int wv = tid >> 5;

  for (int t = 0; t < T_; ++t) {
    const int p = t & 1, pn = p ^ 1;

    // ---- Phase 1: LSTM1 (step t) + output projection (step t-1) ----
    {
      const int njobs = (t > 0) ? (NG / 64 + F_ / 64) : (NG / 64);
      for (int job = wg; job < njobs; job += NWG) {
        if (job < NG / 64) {
          lstm_tile(xh1 + (size_t)p * B_ * KK, W1p, b1p, job * 64, c1,
                    xh1 + (size_t)pn * B_ * KK,   // h1 -> next [ctx,h1]
                    xh2 + (size_t)p  * B_ * KK,   // h1 -> this step's [h1,h2]
                    nullptr, smem, tid);
        } else {
          proj_tile(xho + (size_t)pn * B_ * KK, Wop, bop, (job - NG / 64) * 64,
                    out + (size_t)(t - 1) * F_, tid);
        }
      }
    }
    grid_barrier(bar);

    // ---- Phase 2: LSTM2 ----
    for (int job = wg; job < NG / 64; job += NWG) {
      lstm_tile(xh2 + (size_t)p * B_ * KK, W2p, b2p, job * 64, c2,
                xh2 + (size_t)pn * B_ * KK,       // h2 -> next [h1,h2]
                xho + (size_t)p  * B_ * KK,       // h2 -> this step's [h2,ctx]
                h2f, smem, tid);
    }
    grid_barrier(bar);

    // ---- Phase 3: attention energies: energy[b,tt] = keys[b,tt,:] . h2[b,:] ----
    {
      int gw = wg * (TPB / 32) + wv;
      for (int pr = gw; pr < B_ * T_; pr += NWG * (TPB / 32)) {
        int b = pr >> 8, tt = pr & (T_ - 1);
        const float4* kp = (const float4*)(keys + ((size_t)b * T_ + tt) * KDIM) + lane * 4;
        const float4* hp = (const float4*)(h2f + (size_t)b * KDIM) + lane * 4;
        float acc = 0.f;
        #pragma unroll
        for (int i = 0; i < 4; ++i) {
          float4 kv = kp[i], hv = hp[i];
          acc += kv.x * hv.x + kv.y * hv.y + kv.z * hv.z + kv.w * hv.w;
        }
        acc = wred_sum(acc);
        if (lane == 0) energy[pr] = acc;
      }
    }
    grid_barrier(bar);

    // ---- Phase 4: softmax over T + context = attn @ values ----
    for (int b = wg; b < B_; b += NWG) {
      float e = energy[b * T_ + tid];
      float m = wred_max(e);
      if (lane == 0) sred[wv] = m;
      __syncthreads();
      if (tid == 0) { float mm = sred[0]; for (int i = 1; i < 8; ++i) mm = fmaxf(mm, sred[i]); sbc[0] = mm; }
      __syncthreads();
      float ex = __expf(e - sbc[0]);
      float s = wred_sum(ex);
      if (lane == 0) sred[wv] = s;
      __syncthreads();
      if (tid == 0) { float ss = 0.f; for (int i = 0; i < 8; ++i) ss += sred[i]; sbc[1] = 1.f / ss; }
      __syncthreads();
      sattn[tid] = ex * sbc[1];
      __syncthreads();
      __bf16* ctxA = xh1 + (size_t)pn * B_ * KK + (size_t)b * KK;       // next [ctx,h1], cols 0..511
      __bf16* ctxO = xho + (size_t)p  * B_ * KK + (size_t)b * KK + H_;  // [h2,ctx], cols 512..1023
      {
        int v = tid * 2;  // V_ == 2*TPB -> one float2 per thread, coalesced
        const float2* vp = (const float2*)(values + (size_t)b * T_ * V_ + v);
        float accx = 0.f, accy = 0.f;
        #pragma unroll 4
        for (int tt = 0; tt < T_; ++tt) {
          float2 vv = vp[(size_t)tt * (V_ / 2)];
          accx += sattn[tt] * vv.x;
          accy += sattn[tt] * vv.y;
        }
        __bf16 cx = f2bf(accx), cy = f2bf(accy);
        ctxA[v] = cx; ctxA[v + 1] = cy;
        ctxO[v] = cx; ctxO[v + 1] = cy;
      }
      __syncthreads();
    }
    grid_barrier(bar);
  }

  // ---- final projection for t = T_-1 (reads xho parity of step 255 = 1) ----
  for (int job = wg; job < F_ / 64; job += NWG) {
    proj_tile(xho + (size_t)B_ * KK, Wop, bop, job * 64,
              out + (size_t)(T_ - 1) * F_, tid);
  }
}

// ---------------- host entry ----------------
extern "C" void kernel_launch(void* const* d_in, const int* in_sizes, int n_in,
                              void* d_out, int out_size, void* d_ws, size_t ws_size,
                              hipStream_t stream) {
  (void)in_sizes; (void)n_in; (void)out_size; (void)ws_size;
  const float* keys   = (const float*)d_in[0];
  const float* values = (const float*)d_in[1];
  const float* Wih1   = (const float*)d_in[2];
  const float* Whh1   = (const float*)d_in[3];
  const float* bih1   = (const float*)d_in[4];
  const float* bhh1   = (const float*)d_in[5];
  const float* Wih2   = (const float*)d_in[6];
  const float* Whh2   = (const float*)d_in[7];
  const float* bih2   = (const float*)d_in[8];
  const float* bhh2   = (const float*)d_in[9];
  const float* Wout   = (const float*)d_in[10];
  const float* bout   = (const float*)d_in[11];
  float* out = (float*)d_out;

  char* base = (char*)d_ws;
  size_t o = 0;
  auto take = [&](size_t bytes) -> char* {
    char* r = base + o;
    o = (o + bytes + 255) & ~(size_t)255;
    return r;
  };
  __bf16* W1p = (__bf16*)take((size_t)NG * KK * 2);
  __bf16* W2p = (__bf16*)take((size_t)NG * KK * 2);
  __bf16* Wop = (__bf16*)take((size_t)F_ * KK * 2);
  float*  b1p = (float*)take((size_t)NG * 4);
  float*  b2p = (float*)take((size_t)NG * 4);
  float*  bop = (float*)take((size_t)F_ * 4);
  __bf16* xh1 = (__bf16*)take((size_t)2 * B_ * KK * 2);
  __bf16* xh2 = (__bf16*)take((size_t)2 * B_ * KK * 2);
  __bf16* xho = (__bf16*)take((size_t)2 * B_ * KK * 2);
  float*  c1  = (float*)take((size_t)B_ * H_ * 4);
  float*  c2  = (float*)take((size_t)B_ * KDIM * 4);
  float*  h2f = (float*)take((size_t)B_ * KDIM * 4);
  float*  en  = (float*)take((size_t)B_ * T_ * 4);
  unsigned* bar = (unsigned*)take(256);

  int blkW = (int)(((size_t)NG * KK + TPB - 1) / TPB);
  pack_lstm_kernel<<<blkW, TPB, 0, stream>>>(Wih1, Whh1, bih1, bhh1, W1p, b1p);
  pack_lstm_kernel<<<blkW, TPB, 0, stream>>>(Wih2, Whh2, bih2, bhh2, W2p, b2p);
  pack_out_kernel<<<(int)(((size_t)F_ * KK + TPB - 1) / TPB), TPB, 0, stream>>>(Wout, bout, Wop, bop);
  init_state_kernel<<<(B_ * KK + TPB - 1) / TPB, TPB, 0, stream>>>(values, xh1, xh2, xho, c1, c2, bar);
  decoder_persistent<<<NWG, TPB, 0, stream>>>(keys, values, W1p, W2p, Wop, b1p, b2p, bop,
                                              xh1, xh2, xho, c1, c2, h2f, en, bar, out);
}
